// Metric_42674795053594
// MI455X (gfx1250) — compile-verified, matrix operand order checked by reading, not run on hardware
//
#include <hip/hip_runtime.h>
#include <hip/hip_bf16.h>
#include <math.h>

// ---------------------------------------------------------------------------
// Types for CDNA5 WMMA (wave32). 16x16x32 bf16 -> f32 accumulate.
// ---------------------------------------------------------------------------
typedef __attribute__((ext_vector_type(16))) __bf16 bf16x16;
typedef __attribute__((ext_vector_type(8)))  float  f32x8;

union Frag { bf16x16 v; unsigned u[8]; };

__device__ __forceinline__ unsigned short f2bf(float f) {
    union { float f; unsigned u; } x; x.f = f;
    unsigned r = x.u + 0x7FFFu + ((x.u >> 16) & 1u);   // round-to-nearest-even
    return (unsigned short)(r >> 16);
}
__device__ __forceinline__ float bf2f(unsigned short h) {
    union { unsigned u; float f; } x; x.u = ((unsigned)h) << 16;
    return x.f;
}

// ---------------------------------------------------------------------------
// Fused tile GEMM: out = relu(in[M=16*Mtiles, Kp] @ Wt^T + bias), all bf16 LDS
// in/out, weights pre-transposed in global as bf16 [Nout][Kp].
// Block = 256 threads = 8 waves; each wave owns tiles strided by 8.
// A fragment: lane&15 = M row, half = lane>>4 selects K offset (ISA 7.12.2).
// B fragment: lane&15 = N col; same K striping; Wt is [N][Kp] so each VGPR is
//             one aligned dword load of 2 consecutive bf16.
// D: VGPR r -> row r + 8*half, col = lane&15 (32-bit C/D 16x16 layout).
// ---------------------------------------------------------------------------
__device__ __forceinline__ void mlp_gemm_relu(
    const unsigned short* __restrict__ inb,   // LDS [Mtiles*16][256]
    const unsigned short* __restrict__ wt,    // global bf16 [Nout][Kp]
    const float* __restrict__ bias,           // [Nout]
    unsigned short* __restrict__ outb,        // LDS [Mtiles*16][256]
    int Kp, int Nout, int Mtiles, int validRows, int tid)
{
    const int wave = tid >> 5, lane = tid & 31;
    const int half = lane >> 4, l15 = lane & 15;
    const int Ntiles = Nout >> 4;
    for (int t = wave; t < Mtiles * Ntiles; t += 8) {
        const int mt = t % Mtiles, nt = t / Mtiles;
        const unsigned* arow = (const unsigned*)(inb + (size_t)(mt * 16 + l15) * 256);
        const unsigned* brow = (const unsigned*)(wt + (size_t)(nt * 16 + l15) * (size_t)Kp);
        f32x8 acc = {};
        for (int kt = 0; kt < (Kp >> 5); ++kt) {
            Frag a, bfr;
            const int k0 = kt << 5;
#pragma unroll
            for (int i = 0; i < 8; ++i) {
                const int ko = k0 + ((i & 3) << 1) + ((i >> 2) << 4) + (half << 3);
                a.u[i]   = arow[ko >> 1];
                bfr.u[i] = brow[ko >> 1];
            }
            acc = __builtin_amdgcn_wmma_f32_16x16x32_bf16(
                false, a.v, false, bfr.v, (short)0, acc, false, false);
        }
        const int col = (nt << 4) + l15;
        const float bv = bias[col];
#pragma unroll
        for (int r = 0; r < 8; ++r) {
            const int rl = (mt << 4) + r + (half << 3);
            float v = fmaxf(acc[r] + bv, 0.f);
            if (rl >= validRows) v = 0.f;            // keep padded rows exactly 0
            outb[(size_t)rl * 256 + col] = f2bf(v);
        }
    }
}

// Final g-layer: relu(in @ Wt^T + bias) summed over valid rows into colsum[256].
__device__ __forceinline__ void mlp_gemm_relu_sum(
    const unsigned short* __restrict__ inb,
    const unsigned short* __restrict__ wt,    // bf16 [256][Kp]
    const float* __restrict__ bias,
    float* __restrict__ colsum,               // LDS [256]
    int Kp, int Mtiles, int validRows, int tid)
{
    const int wave = tid >> 5, lane = tid & 31;
    const int half = lane >> 4, l15 = lane & 15;
    for (int t = wave; t < Mtiles * 16; t += 8) {
        const int mt = t % Mtiles, nt = t / Mtiles;
        const unsigned* arow = (const unsigned*)(inb + (size_t)(mt * 16 + l15) * 256);
        const unsigned* brow = (const unsigned*)(wt + (size_t)(nt * 16 + l15) * (size_t)Kp);
        f32x8 acc = {};
        for (int kt = 0; kt < (Kp >> 5); ++kt) {
            Frag a, bfr;
            const int k0 = kt << 5;
#pragma unroll
            for (int i = 0; i < 8; ++i) {
                const int ko = k0 + ((i & 3) << 1) + ((i >> 2) << 4) + (half << 3);
                a.u[i]   = arow[ko >> 1];
                bfr.u[i] = brow[ko >> 1];
            }
            acc = __builtin_amdgcn_wmma_f32_16x16x32_bf16(
                false, a.v, false, bfr.v, (short)0, acc, false, false);
        }
        const int col = (nt << 4) + l15;
        const float bv = bias[col];
#pragma unroll
        for (int r = 0; r < 8; ++r) {
            const int rl = (mt << 4) + r + (half << 3);
            if (rl < validRows)
                atomicAdd(&colsum[col], fmaxf(acc[r] + bv, 0.f));  // ds_add_f32
        }
    }
}

// ---------------------------------------------------------------------------
// repnet kernels (direct conv, fp32 VALU — tiny vs the relation GEMM).
// ---------------------------------------------------------------------------
__global__ void k_conv1_pool(const float* __restrict__ x, const float* __restrict__ w,
                             const float* __restrict__ bias, float* __restrict__ out, int N)
{
    // x:[N,3,84,84] -> conv3x3 valid (82) -> maxpool2 -> out:[N,64,41,41]
    int idx = blockIdx.x * blockDim.x + threadIdx.x;
    int total = N * 64 * 41 * 41;
    if (idx >= total) return;
    int ox = idx % 41; int t = idx / 41;
    int oy = t % 41;   t /= 41;
    int oc = t % 64;   int n = t / 64;
    const float* xn = x + (size_t)n * 3 * 84 * 84;
    const float* wc = w + oc * 27;
    float m = -1e30f;
    for (int dy = 0; dy < 2; ++dy)
        for (int dx = 0; dx < 2; ++dx) {
            int cy = 2 * oy + dy, cx = 2 * ox + dx;
            float acc = bias[oc];
            for (int ic = 0; ic < 3; ++ic)
#pragma unroll
                for (int ky = 0; ky < 3; ++ky) {
                    const float* row = xn + ((size_t)(ic * 84) + (cy + ky)) * 84 + cx;
                    const float* wr  = wc + (ic * 3 + ky) * 3;
                    acc += row[0] * wr[0] + row[1] * wr[1] + row[2] * wr[2];
                }
            m = fmaxf(m, acc);
        }
    out[idx] = m;
}

__global__ void k_conv2_pool(const float* __restrict__ x, const float* __restrict__ w,
                             const float* __restrict__ bias, float* __restrict__ out, int N)
{
    // x:[N,64,41,41] -> conv3x3 (39) -> maxpool2 -> out:[N,64,19,19]
    int idx = blockIdx.x * blockDim.x + threadIdx.x;
    int total = N * 64 * 19 * 19;
    if (idx >= total) return;
    int ox = idx % 19; int t = idx / 19;
    int oy = t % 19;   t /= 19;
    int oc = t % 64;   int n = t / 64;
    const float* xn = x + (size_t)n * 64 * 41 * 41;
    float m = -1e30f;
    for (int dy = 0; dy < 2; ++dy)
        for (int dx = 0; dx < 2; ++dx) {
            int cy = 2 * oy + dy, cx = 2 * ox + dx;
            float acc = bias[oc];
            for (int ic = 0; ic < 64; ++ic)
#pragma unroll
                for (int ky = 0; ky < 3; ++ky) {
                    const float* row = xn + ((size_t)(ic * 41) + (cy + ky)) * 41 + cx;
                    const float* wr  = w + (((size_t)oc * 64 + ic) * 3 + ky) * 3;
                    acc += row[0] * wr[0] + row[1] * wr[1] + row[2] * wr[2];
                }
            m = fmaxf(m, acc);
        }
    out[idx] = m;
}

__global__ void k_conv64(const float* __restrict__ x, const float* __restrict__ w,
                         const float* __restrict__ bias, float* __restrict__ out,
                         int N, int H)
{
    // x:[N,64,H,H] -> conv3x3 valid -> out:[N,64,H-2,H-2]
    int HO = H - 2;
    int idx = blockIdx.x * blockDim.x + threadIdx.x;
    int total = N * 64 * HO * HO;
    if (idx >= total) return;
    int ox = idx % HO; int t = idx / HO;
    int oy = t % HO;   t /= HO;
    int oc = t % 64;   int n = t / 64;
    const float* xn = x + (size_t)n * 64 * H * H;
    float acc = bias[oc];
    for (int ic = 0; ic < 64; ++ic)
#pragma unroll
        for (int ky = 0; ky < 3; ++ky) {
            const float* row = xn + ((size_t)(ic * H) + (oy + ky)) * H + ox;
            const float* wr  = w + (((size_t)oc * 64 + ic) * 3 + ky) * 3;
            acc += row[0] * wr[0] + row[1] * wr[1] + row[2] * wr[2];
        }
    out[idx] = acc;
}

__global__ void k_bn2d_stats(const float* __restrict__ x, int N, int HW,
                             float* __restrict__ stat)   // stat[c]=mean, stat[64+c]=rstd
{
    __shared__ float ss[256], sq[256];
    int c = blockIdx.x;
    int tot = N * HW;
    float s = 0.f, q = 0.f;
    for (int i = threadIdx.x; i < tot; i += 256) {
        int n = i / HW, r = i - n * HW;
        float v = x[((size_t)n * 64 + c) * HW + r];
        s += v; q += v * v;
    }
    ss[threadIdx.x] = s; sq[threadIdx.x] = q;
    __syncthreads();
    for (int st = 128; st > 0; st >>= 1) {
        if ((int)threadIdx.x < st) { ss[threadIdx.x] += ss[threadIdx.x + st];
                                     sq[threadIdx.x] += sq[threadIdx.x + st]; }
        __syncthreads();
    }
    if (threadIdx.x == 0) {
        float m = ss[0] / (float)tot;
        float var = sq[0] / (float)tot - m * m;
        stat[c] = m;
        stat[64 + c] = rsqrtf(var + 1e-5f);
    }
}

__global__ void k_bn2d_apply_relu(float* __restrict__ x, int N, int HW,
                                  const float* __restrict__ stat,
                                  const float* __restrict__ g,
                                  const float* __restrict__ b)
{
    int idx = blockIdx.x * blockDim.x + threadIdx.x;
    int total = N * 64 * HW;
    if (idx >= total) return;
    int c = (idx / HW) % 64;
    float v = (x[idx] - stat[c]) * stat[64 + c] * g[c] + b[c];
    x[idx] = fmaxf(v, 0.f);
}

__global__ void k_avgpool_feat(const float* __restrict__ x, float* __restrict__ feat, int N)
{
    // x:[N,64,15,15] -> feat:[N,9,66]  (ch 0..63 = 5x5 mean, 64 = i/3, 65 = j/3)
    int idx = blockIdx.x * blockDim.x + threadIdx.x;
    int total = N * 9 * 66;
    if (idx >= total) return;
    int ch = idx % 66; int t = idx / 66;
    int pos = t % 9;   int n = t / 9;
    int i = pos / 3, j = pos % 3;
    float v;
    if (ch < 64) {
        const float* p = x + (((size_t)n * 64 + ch) * 225) + (i * 5) * 15 + j * 5;
        float s = 0.f;
#pragma unroll
        for (int r = 0; r < 5; ++r)
#pragma unroll
            for (int c = 0; c < 5; ++c) s += p[r * 15 + c];
        v = s * (1.f / 25.f);
    } else {
        v = (float)(ch == 64 ? i : j) * (1.f / 3.f);
    }
    feat[idx] = v;
}

// Transpose+convert f32 W[K][Nn] -> bf16 Wt[Nn][Kp] (K padded with zeros).
__global__ void k_prep_wt(const float* __restrict__ w, unsigned short* __restrict__ wt,
                          int K, int Nn, int Kp)
{
    int idx = blockIdx.x * blockDim.x + threadIdx.x;
    int total = Nn * Kp;
    if (idx >= total) return;
    int k = idx % Kp, n = idx / Kp;
    float v = (k < K) ? w[(size_t)k * Nn + n] : 0.f;
    wt[idx] = f2bf(v);
}

__global__ void k_zero_f32(float* __restrict__ p, int n)
{
    int idx = blockIdx.x * blockDim.x + threadIdx.x;
    if (idx < n) p[idx] = 0.f;
}

// ---------------------------------------------------------------------------
// Fused relation g-MLP. One block per (b,q,s,chunk): builds a 32-row slice of
// the 81x132 pair matrix in LDS (bf16), runs 4 WMMA GEMM layers fused, and
// column-sum-accumulates relu(layer4) into xf[bqs][256] with global atomics.
// ---------------------------------------------------------------------------
__global__ void __launch_bounds__(256)
k_gmlp(const float* __restrict__ featS,   // [40][9][66]
       const float* __restrict__ featQ,   // [240][9][66]
       const unsigned short* __restrict__ gw1t,  // bf16 [256][160]
       const unsigned short* __restrict__ gw2t,  // bf16 [256][256]
       const unsigned short* __restrict__ gw3t,
       const unsigned short* __restrict__ gw4t,
       const float* __restrict__ gb1, const float* __restrict__ gb2,
       const float* __restrict__ gb3, const float* __restrict__ gb4,
       float* __restrict__ xf)            // [1200][256], pre-zeroed
{
    __shared__ unsigned short bufA[32 * 256];
    __shared__ unsigned short bufB[32 * 256];
    __shared__ float colsum[256];

    const int tid = threadIdx.x;
    const int bid = blockIdx.x;            // 0..3599
    const int bqs = bid / 3, chunk = bid % 3;
    const int s = bqs % 5, q = (bqs / 5) % 30, b = bqs / 150;
    const int rowBase = chunk * 32;
    int vr = 81 - rowBase; if (vr > 32) vr = 32;   // valid rows in this chunk

    if (tid < 256) colsum[tid] = 0.f;

    // Warm next-layer weights into cache (global_prefetch_b8).
    if (tid < 64) {
        __builtin_prefetch(gw2t + (size_t)tid * 1024, 0, 0);
        __builtin_prefetch(gw3t + (size_t)tid * 1024, 0, 0);
        __builtin_prefetch(gw4t + (size_t)tid * 1024, 0, 0);
    }

    // Build pair slice: row r = x*9+y -> [supFeat(b,s,x) | qryFeat(b,q,y) | 0pad]
    for (int e = tid; e < 32 * 160; e += 256) {
        int k = e % 160, rl = e / 160;
        int r = rowBase + rl;
        float v = 0.f;
        if (r < 81 && k < 132) {
            int px = r / 9, py = r % 9;
            if (k < 66) v = featS[(((size_t)b * 5 + s) * 9 + px) * 66 + k];
            else        v = featQ[(((size_t)b * 30 + q) * 9 + py) * 66 + (k - 66)];
        }
        bufA[rl * 256 + k] = f2bf(v);
    }
    __syncthreads();

    mlp_gemm_relu(bufA, gw1t, gb1, bufB, 160, 256, 2, vr, tid);
    __syncthreads();
    mlp_gemm_relu(bufB, gw2t, gb2, bufA, 256, 256, 2, vr, tid);
    __syncthreads();
    mlp_gemm_relu(bufA, gw3t, gb3, bufB, 256, 256, 2, vr, tid);
    __syncthreads();
    mlp_gemm_relu_sum(bufB, gw4t, gb4, colsum, 256, 2, vr, tid);
    __syncthreads();

    if (tid < 256) atomicAdd(&xf[(size_t)bqs * 256 + tid], colsum[tid]);
}

// BN1d over the 1200 rows: ab[c] = g*rstd, ab[256+c] = b - mean*g*rstd
__global__ void k_bn1d_stats(const float* __restrict__ xf,
                             const float* __restrict__ g, const float* __restrict__ b,
                             float* __restrict__ ab)
{
    int c = threadIdx.x;
    float s = 0.f, ss = 0.f;
    for (int r = 0; r < 1200; ++r) {
        float v = xf[(size_t)r * 256 + c];
        s += v; ss += v * v;
    }
    float m = s * (1.f / 1200.f);
    float var = ss * (1.f / 1200.f) - m * m;
    float a = g[c] * rsqrtf(var + 1e-5f);
    ab[c] = a;
    ab[256 + c] = b[c] - m * a;
}

// f-MLP: 16 rows per block, WMMA layers 256->256->256->64, then 64->1 dot,
// sigmoid, store dist = 1 - score.
__global__ void __launch_bounds__(256)
k_fmlp(const float* __restrict__ xf, const float* __restrict__ ab,
       const unsigned short* __restrict__ fw1t, const float* __restrict__ fb1,
       const unsigned short* __restrict__ fw2t, const float* __restrict__ fb2,
       const unsigned short* __restrict__ fw3t, const float* __restrict__ fb3,
       const float* __restrict__ fw4, const float* __restrict__ fb4,
       float* __restrict__ dist)
{
    __shared__ unsigned short bufA[16 * 256];
    __shared__ unsigned short bufB[16 * 256];
    const int tid = threadIdx.x;
    const int row0 = blockIdx.x * 16;

    for (int e = tid; e < 16 * 256; e += 256) {
        int c = e & 255, r = e >> 8;
        float v = xf[(size_t)(row0 + r) * 256 + c] * ab[c] + ab[256 + c];
        bufA[e] = f2bf(v);
    }
    __syncthreads();
    mlp_gemm_relu(bufA, fw1t, fb1, bufB, 256, 256, 1, 16, tid);
    __syncthreads();
    mlp_gemm_relu(bufB, fw2t, fb2, bufA, 256, 256, 1, 16, tid);
    __syncthreads();
    mlp_gemm_relu(bufA, fw3t, fb3, bufB, 256, 64, 1, 16, tid);
    __syncthreads();
    if (tid < 16) {
        float s = fb4[0];
        for (int k = 0; k < 64; ++k) s += bf2f(bufB[tid * 256 + k]) * fw4[k];
        float sig = 1.f / (1.f + expf(-s));
        dist[row0 + tid] = 1.f - sig;
    }
}

// Margin loss: per (b,q) pick positive dist and min negative dist; sum/b.
__global__ void k_loss(const float* __restrict__ dist,
                       const int* __restrict__ sup_y, const int* __restrict__ qry_y,
                       float* __restrict__ out)
{
    __shared__ float sh[256];
    int tid = threadIdx.x;
    float v = 0.f;
    if (tid < 240) {
        int b = tid / 30, q = tid % 30;
        int qy = qry_y[b * 30 + q];
        const float* d = dist + (size_t)(b * 30 + q) * 5;
        float dap = 0.f, mn = 1e30f;
        for (int s = 0; s < 5; ++s) {
            float dd = d[s];
            if (sup_y[b * 5 + s] == qy) dap = dd;
            else mn = fminf(mn, dd);
        }
        v = fmaxf(dap - mn + 0.2f, 0.f);
    }
    sh[tid] = v;
    __syncthreads();
    for (int st = 128; st > 0; st >>= 1) {
        if (tid < st) sh[tid] += sh[tid + st];
        __syncthreads();
    }
    if (tid == 0) out[0] = sh[0] * (1.f / 8.f);
}

// ---------------------------------------------------------------------------
// Host launch
// ---------------------------------------------------------------------------
extern "C" void kernel_launch(void* const* d_in, const int* in_sizes, int n_in,
                              void* d_out, int out_size, void* d_ws, size_t ws_size,
                              hipStream_t stream)
{
    (void)in_sizes; (void)n_in; (void)out_size; (void)ws_size;

    const float* support_x = (const float*)d_in[0];
    const float* query_x   = (const float*)d_in[1];
    const int*   support_y = (const int*)d_in[2];
    const int*   query_y   = (const int*)d_in[3];
    const float* w1 = (const float*)d_in[4];
    const float* w2 = (const float*)d_in[5];
    const float* w3 = (const float*)d_in[6];
    const float* w4 = (const float*)d_in[7];
    const float* b1 = (const float*)d_in[8];
    const float* bn1_g = (const float*)d_in[9];
    const float* bn1_b = (const float*)d_in[10];
    const float* b2 = (const float*)d_in[11];
    const float* bn2_g = (const float*)d_in[12];
    const float* bn2_b = (const float*)d_in[13];
    const float* b3 = (const float*)d_in[14];
    const float* bn3_g = (const float*)d_in[15];
    const float* bn3_b = (const float*)d_in[16];
    const float* b4 = (const float*)d_in[17];
    const float* bn4_g = (const float*)d_in[18];
    const float* bn4_b = (const float*)d_in[19];
    const float* gw1 = (const float*)d_in[20];
    const float* gw2 = (const float*)d_in[21];
    const float* gw3 = (const float*)d_in[22];
    const float* gw4 = (const float*)d_in[23];
    const float* gb1 = (const float*)d_in[24];
    const float* gb2 = (const float*)d_in[25];
    const float* gb3 = (const float*)d_in[26];
    const float* gb4 = (const float*)d_in[27];
    const float* fbn_g = (const float*)d_in[28];
    const float* fbn_b = (const float*)d_in[29];
    const float* fw1 = (const float*)d_in[30];
    const float* fb1 = (const float*)d_in[31];
    const float* fw2 = (const float*)d_in[32];
    const float* fb2 = (const float*)d_in[33];
    const float* fw3 = (const float*)d_in[34];
    const float* fb3 = (const float*)d_in[35];
    const float* fw4 = (const float*)d_in[36];
    const float* fb4 = (const float*)d_in[37];

    // Workspace carve (bump allocator, 256B aligned).
    char* base = (char*)d_ws; size_t off = 0;
    auto carve = [&](size_t bytes) -> void* {
        void* p = base + off;
        off = (off + bytes + 255) & ~(size_t)255;
        return p;
    };
    float* BUF1  = (float*)carve((size_t)240 * 64 * 41 * 41 * 4);
    float* BUF2  = (float*)carve((size_t)240 * 64 * 19 * 19 * 4);
    float* BUF3  = (float*)carve((size_t)240 * 64 * 17 * 17 * 4);
    float* BUF4  = (float*)carve((size_t)240 * 64 * 15 * 15 * 4);
    float* FEATS = (float*)carve((size_t)40  * 9 * 66 * 4);
    float* FEATQ = (float*)carve((size_t)240 * 9 * 66 * 4);
    float* XF    = (float*)carve((size_t)1200 * 256 * 4);
    float* AB    = (float*)carve(512 * 4);
    float* STAT  = (float*)carve(128 * 4);
    float* DIST  = (float*)carve(1200 * 4);
    unsigned short* GW1T = (unsigned short*)carve((size_t)256 * 160 * 2);
    unsigned short* GW2T = (unsigned short*)carve((size_t)256 * 256 * 2);
    unsigned short* GW3T = (unsigned short*)carve((size_t)256 * 256 * 2);
    unsigned short* GW4T = (unsigned short*)carve((size_t)256 * 256 * 2);
    unsigned short* FW1T = (unsigned short*)carve((size_t)256 * 256 * 2);
    unsigned short* FW2T = (unsigned short*)carve((size_t)256 * 256 * 2);
    unsigned short* FW3T = (unsigned short*)carve((size_t)64  * 256 * 2);

    const int T = 256;
    auto blocks = [](int total) { return (total + 255) / 256; };

    // Weight prep (transpose + bf16) — independent of images.
    k_prep_wt<<<blocks(256 * 160), T, 0, stream>>>(gw1, GW1T, 132, 256, 160);
    k_prep_wt<<<blocks(256 * 256), T, 0, stream>>>(gw2, GW2T, 256, 256, 256);
    k_prep_wt<<<blocks(256 * 256), T, 0, stream>>>(gw3, GW3T, 256, 256, 256);
    k_prep_wt<<<blocks(256 * 256), T, 0, stream>>>(gw4, GW4T, 256, 256, 256);
    k_prep_wt<<<blocks(256 * 256), T, 0, stream>>>(fw1, FW1T, 256, 256, 256);
    k_prep_wt<<<blocks(256 * 256), T, 0, stream>>>(fw2, FW2T, 256, 256, 256);
    k_prep_wt<<<blocks(64  * 256), T, 0, stream>>>(fw3, FW3T, 256, 64, 256);
    k_zero_f32<<<blocks(1200 * 256), T, 0, stream>>>(XF, 1200 * 256);

    // repnet: support batch (N=40) and query batch (N=240) get SEPARATE BN stats.
    struct Grp { int N; const float* x; float* feat; };
    Grp grp[2] = { {40, support_x, FEATS}, {240, query_x, FEATQ} };
    for (int g = 0; g < 2; ++g) {
        const int N = grp[g].N;
        int tot1 = N * 64 * 41 * 41;
        k_conv1_pool<<<blocks(tot1), T, 0, stream>>>(grp[g].x, w1, b1, BUF1, N);
        k_bn2d_stats<<<64, T, 0, stream>>>(BUF1, N, 1681, STAT);
        k_bn2d_apply_relu<<<blocks(tot1), T, 0, stream>>>(BUF1, N, 1681, STAT, bn1_g, bn1_b);
        int tot2 = N * 64 * 19 * 19;
        k_conv2_pool<<<blocks(tot2), T, 0, stream>>>(BUF1, w2, b2, BUF2, N);
        k_bn2d_stats<<<64, T, 0, stream>>>(BUF2, N, 361, STAT);
        k_bn2d_apply_relu<<<blocks(tot2), T, 0, stream>>>(BUF2, N, 361, STAT, bn2_g, bn2_b);
        int tot3 = N * 64 * 17 * 17;
        k_conv64<<<blocks(tot3), T, 0, stream>>>(BUF2, w3, b3, BUF3, N, 19);
        k_bn2d_stats<<<64, T, 0, stream>>>(BUF3, N, 289, STAT);
        k_bn2d_apply_relu<<<blocks(tot3), T, 0, stream>>>(BUF3, N, 289, STAT, bn3_g, bn3_b);
        int tot4 = N * 64 * 15 * 15;
        k_conv64<<<blocks(tot4), T, 0, stream>>>(BUF3, w4, b4, BUF4, N, 17);
        k_bn2d_stats<<<64, T, 0, stream>>>(BUF4, N, 225, STAT);
        k_bn2d_apply_relu<<<blocks(tot4), T, 0, stream>>>(BUF4, N, 225, STAT, bn4_g, bn4_b);
        int totf = N * 9 * 66;
        k_avgpool_feat<<<blocks(totf), T, 0, stream>>>(BUF4, grp[g].feat, N);
    }

    // Fused relation MLP on WMMA: 1200 (b,q,s) triples x 3 row-chunks.
    k_gmlp<<<3600, T, 0, stream>>>(FEATS, FEATQ, GW1T, GW2T, GW3T, GW4T,
                                   gb1, gb2, gb3, gb4, XF);

    // BN1d + f-MLP + loss.
    k_bn1d_stats<<<1, T, 0, stream>>>(XF, fbn_g, fbn_b, AB);
    k_fmlp<<<75, T, 0, stream>>>(XF, AB, FW1T, fb1, FW2T, fb2, FW3T, fb3,
                                 fw4, fb4, DIST);
    k_loss<<<1, T, 0, stream>>>(DIST, support_y, query_y, (float*)d_out);
}
